// GraphicallyStructuredModel_16887811407980
// MI455X (gfx1250) — compile-verified, matrix operand order checked by reading, not run on hardware
//
#include <hip/hip_runtime.h>
#include <hip/hip_bf16.h>
#include <math.h>

typedef __attribute__((ext_vector_type(16))) _Float16 v16h;
typedef __attribute__((ext_vector_type(8)))  float    v8f;
typedef __attribute__((ext_vector_type(4)))  unsigned int u32x4;
typedef __attribute__((ext_vector_type(8)))  int i32x8;
typedef __attribute__((ext_vector_type(4)))  int i32x4;

#define BATCH   2
#define CDIM    512
#define NPTS    1024
#define KNBR    64
#define NHEADS  8
#define DHEAD   64
#define NGROUPS 32
#define CPG     (CDIM / NGROUPS)   // 16

#if __has_builtin(__builtin_amdgcn_tensor_load_to_lds)
#define USE_TDM 1
#else
#define USE_TDM 0
#endif

// ---------------------------------------------------------------------------
// GroupNorm: one block per (group, batch). Writes output directly as f16,
// transposed to [b][n][c] — the exact row-major-K layout the GEMM B operand
// (and the TDM tile descriptor) wants.
// ---------------------------------------------------------------------------
__global__ __launch_bounds__(256) void gn_kernel(const float* __restrict__ x,
                                                 const float* __restrict__ gamma,
                                                 const float* __restrict__ beta,
                                                 _Float16* __restrict__ hT) {
  int g = blockIdx.x;           // group 0..31
  int b = blockIdx.y;           // batch 0..1
  const int total = CPG * NPTS; // 16384
  const float* xp = x + ((size_t)b * CDIM + g * CPG) * NPTS;

  float s = 0.f, ss = 0.f;
  for (int i = threadIdx.x; i < total; i += 256) {
    float vv = xp[i];
    s += vv;
    ss += vv * vv;
  }
  __shared__ float red0[256], red1[256];
  red0[threadIdx.x] = s;
  red1[threadIdx.x] = ss;
  __syncthreads();
  for (int off = 128; off > 0; off >>= 1) {
    if (threadIdx.x < off) {
      red0[threadIdx.x] += red0[threadIdx.x + off];
      red1[threadIdx.x] += red1[threadIdx.x + off];
    }
    __syncthreads();
  }
  float mu  = red0[0] / (float)total;
  float var = red1[0] / (float)total - mu * mu;
  float rsig = rsqrtf(var + 1e-6f);
  for (int i = threadIdx.x; i < total; i += 256) {
    int lc = i / NPTS;                 // 0..15 channel-in-group
    int j  = i - lc * NPTS;            // node index
    int cc = g * CPG + lc;             // absolute channel
    float val = (xp[i] - mu) * rsig * gamma[cc] + beta[cc];
    hT[((size_t)b * NPTS + j) * CDIM + cc] = (_Float16)val;
  }
}

// ---------------------------------------------------------------------------
// f32 -> f16 weight conversion (one-shot, removes conversion from GEMM path)
// ---------------------------------------------------------------------------
__global__ __launch_bounds__(256) void cvt_kernel(const float* __restrict__ src,
                                                  _Float16* __restrict__ dst,
                                                  int n) {
  int i = blockIdx.x * 256 + threadIdx.x;
  if (i < n) dst[i] = (_Float16)src[i];
}

// ---------------------------------------------------------------------------
// WMMA GEMM: C[b] = W16[M,K] * BT16[b][N,K]^T + bias (+ optional residual).
// Block tile 128x128, 8 waves each computing 64x32 via 4x2 16x16 WMMA tiles.
// Tiles staged into LDS by the Tensor Data Mover (pad_enable reproduces the
// 40-half row stride used by the fragment gathers).
// ---------------------------------------------------------------------------
#define TM 128
#define TN 128
#define TK 32
#define LDT (TK + 8)   // 32 halfs data + TDM pad of 4 DWORDs per 16 DWORDs

#if USE_TDM
// Pack a 2D Tensor-DMA descriptor (ISA 8.3-8.6) and issue TENSOR_LOAD_TO_LDS.
// Tile: tile_d1 rows x tile_d0 elements (2-byte), row stride = stride0 elems.
// This toolchain exposes the 6-arg builtin:
//   (u32x4 g0, i32x8 g1, i32x4 g2, i32x4 g3, i32x8 gx, i32 cpol)
__device__ __forceinline__ void tdm_load_2d(unsigned long long ga, unsigned lds,
                                            unsigned tensor_d0, unsigned tensor_d1,
                                            unsigned tile_d0, unsigned tile_d1,
                                            unsigned long long stride0) {
  u32x4 g0;
  g0[0] = 1u;                                         // count=1, user D#
  g0[1] = lds;                                        // lds_addr
  g0[2] = (unsigned)(ga & 0xFFFFFFFFull);             // global_addr[31:0]
  g0[3] = (unsigned)((ga >> 32) & 0x01FFFFFFull)      // global_addr[56:32]
        | (2u << 30);                                 // type = 2 ("image")
  i32x8 g1;
  g1[0] = (1 << 16)                                   // data_size = 2 bytes
        | (1 << 20)                                   // pad_enable
        | (3 << 22)                                   // pad_interval: 16 DWORDs
        | (3 << 25);                                  // pad_amount: 4 DWORDs
  g1[1] = (int)((tensor_d0 & 0xFFFFu) << 16);                 // dim0[15:0]
  g1[2] = (int)(((tensor_d0 >> 16) & 0xFFFFu) |
                ((tensor_d1 & 0xFFFFu) << 16));               // dim0[31:16]|dim1[15:0]
  g1[3] = (int)(((tensor_d1 >> 16) & 0xFFFFu) |
                ((tile_d0 & 0xFFFFu) << 16));                 // dim1[31:16]|tile0
  g1[4] = (int)(tile_d1 & 0xFFFFu);                           // tile1, tile2=0
  g1[5] = (int)(stride0 & 0xFFFFFFFFull);                     // stride0[31:0]
  g1[6] = (int)((stride0 >> 32) & 0xFFFFull);                 // stride0[47:32]
  g1[7] = 0;
  i32x4 gz4 = {0, 0, 0, 0};
  i32x8 gz8 = {0, 0, 0, 0, 0, 0, 0, 0};
  __builtin_amdgcn_tensor_load_to_lds(g0, g1, gz4, gz4, gz8, 0);
}
#endif

// Per-lane fragment gather matching the CDNA5 16-bit A-matrix 16x32 layout
// (ISA 7.12.2). B is stored [N][K] so the same gather applies (lane = N).
__device__ __forceinline__ v16h load_frag(const _Float16* base, int lane) {
  const _Float16* p = base + (lane & 15) * LDT + ((lane & 16) ? 8 : 0);
  v16h r;
#pragma unroll
  for (int vv = 0; vv < 4; ++vv) {
    r[2 * vv]         = p[2 * vv];
    r[2 * vv + 1]     = p[2 * vv + 1];
    r[8 + 2 * vv]     = p[16 + 2 * vv];
    r[8 + 2 * vv + 1] = p[16 + 2 * vv + 1];
  }
  return r;
}

__global__ __launch_bounds__(256) void gemm_kernel(
    const _Float16* __restrict__ W16,   // [M][K] row-major (f16)
    const float* __restrict__ bias,     // [M]
    const _Float16* __restrict__ BT16,  // [batch][N][K] row-major (f16)
    const float* __restrict__ resid,    // [batch][M][N] or nullptr
    float* __restrict__ C,              // [batch][M][N]
    int M, int N, int Kd) {
  __shared__ __align__(16) _Float16 As[TM][LDT];
  __shared__ __align__(16) _Float16 Bs[TN][LDT];

  int bz = blockIdx.z;
  const _Float16* BTb = BT16 + (size_t)bz * N * Kd;
  float* Cb = C + (size_t)bz * M * N;
  const float* Rb = resid ? resid + (size_t)bz * M * N : nullptr;

  int m0 = blockIdx.x * TM;
  int n0 = blockIdx.y * TN;
  int tid = threadIdx.x;
  int wave = tid >> 5, lane = tid & 31;
  int wm = (wave & 1) * 64;   // wave M offset within block tile
  int wn = (wave >> 1) * 32;  // wave N offset within block tile

  v8f acc[4][2] = {};

  for (int kk = 0; kk < Kd; kk += TK) {
#if USE_TDM
    if (tid < 32) {  // wave 0 drives the DMA; TDM ignores EXEC, per-wave issue
      tdm_load_2d((unsigned long long)(size_t)(W16 + (size_t)m0 * Kd + kk),
                  (unsigned)(size_t)&As[0][0],
                  (unsigned)Kd, (unsigned)TM, TK, TM, (unsigned)Kd);
      tdm_load_2d((unsigned long long)(size_t)(BTb + (size_t)n0 * Kd + kk),
                  (unsigned)(size_t)&Bs[0][0],
                  (unsigned)Kd, (unsigned)TN, TK, TN, (unsigned)Kd);
      __builtin_amdgcn_s_wait_tensorcnt(0);
    }
#else
    {
      int row = tid >> 1;
      int kh = (tid & 1) * 16;
      const _Float16* srcA = W16 + (size_t)(m0 + row) * Kd + kk + kh;
      const _Float16* srcB = BTb + (size_t)(n0 + row) * Kd + kk + kh;
#pragma unroll
      for (int i = 0; i < 16; ++i) {
        As[row][kh + i] = srcA[i];
        Bs[row][kh + i] = srcB[i];
      }
    }
#endif
    __syncthreads();
#pragma unroll
    for (int i = 0; i < 4; ++i) {
      v16h a = load_frag(&As[wm + i * 16][0], lane);
#pragma unroll
      for (int j = 0; j < 2; ++j) {
        v16h bb = load_frag(&Bs[wn + j * 16][0], lane);
        acc[i][j] = __builtin_amdgcn_wmma_f32_16x16x32_f16(
            false, a, false, bb, (short)0, acc[i][j], false, false);
      }
    }
    __syncthreads();
  }

  // Epilogue: C layout (ISA 7.12.2): lane&15 = N, VGPR r + 8*lane[4] = M.
  int nlo = lane & 15;
  int mhi = (lane & 16) ? 8 : 0;
#pragma unroll
  for (int i = 0; i < 4; ++i) {
#pragma unroll
    for (int j = 0; j < 2; ++j) {
      int ncol = n0 + wn + j * 16 + nlo;
#pragma unroll
      for (int r = 0; r < 8; ++r) {
        int mrow = m0 + wm + i * 16 + r + mhi;
        float val = acc[i][j][r] + bias[mrow];
        if (Rb) val += Rb[(size_t)mrow * N + ncol];
        Cb[(size_t)mrow * N + ncol] = val;
      }
    }
  }
}

// ---------------------------------------------------------------------------
// Gathered sparse attention: one thread per (b, head, query).
// q loads coalesce across threads (consecutive j); k/v gathers hit L2.
// Output written as f16 transposed [b][n][c] with c = d*H + h (the permuted
// channel layout), ready to be the B operand of the output-projection GEMM.
// ---------------------------------------------------------------------------
__global__ __launch_bounds__(256) void attn_kernel(
    const float* __restrict__ q, const float* __restrict__ k,
    const float* __restrict__ v, const long long* __restrict__ idx,
    const int* __restrict__ mask, _Float16* __restrict__ oT) {
  int t = blockIdx.x * 256 + threadIdx.x;  // 0..16383
  int j = t & (NPTS - 1);
  int bh = t >> 10;
  int b = bh >> 3;
  int hh = bh & 7;
  size_t base = ((size_t)b * CDIM + hh * DHEAD) * NPTS;

  float qv[DHEAD];
#pragma unroll
  for (int d = 0; d < DHEAD; ++d) qv[d] = q[base + (size_t)d * NPTS + j];

  float s[KNBR];
  float mx = -INFINITY;
  for (int kk = 0; kk < KNBR; ++kk) {
    int id = (int)idx[(size_t)j * KNBR + kk];
    const float* kp = k + base + id;
    float acc = 0.f;
#pragma unroll
    for (int d = 0; d < DHEAD; ++d) acc += qv[d] * kp[(size_t)d * NPTS];
    s[kk] = (mask[j * KNBR + kk] != 0) ? acc : -INFINITY;
    mx = fmaxf(mx, s[kk]);
  }
  float sum = 0.f;
  for (int kk = 0; kk < KNBR; ++kk) {
    float e = expf(s[kk] - mx);  // masked entries: exp(-inf)=0
    s[kk] = e;
    sum += e;
  }
  float inv = 1.f / sum;

  float ov[DHEAD];
#pragma unroll
  for (int d = 0; d < DHEAD; ++d) ov[d] = 0.f;
  for (int kk = 0; kk < KNBR; ++kk) {
    float w = s[kk] * inv;
    int id = (int)idx[(size_t)j * KNBR + kk];
    const float* vp = v + base + id;
#pragma unroll
    for (int d = 0; d < DHEAD; ++d) ov[d] += w * vp[(size_t)d * NPTS];
  }
  _Float16* op = oT + ((size_t)b * NPTS + j) * CDIM;
#pragma unroll
  for (int d = 0; d < DHEAD; ++d)
    op[d * NHEADS + hh] = (_Float16)ov[d];
}

// ---------------------------------------------------------------------------
extern "C" void kernel_launch(void* const* d_in, const int* in_sizes, int n_in,
                              void* d_out, int out_size, void* d_ws,
                              size_t ws_size, hipStream_t stream) {
  const float* x           = (const float*)d_in[0];
  const int* valid_mask    = (const int*)d_in[1];
  const long long* att_idx = (const long long*)d_in[2];
  const float* gn_gamma    = (const float*)d_in[3];
  const float* gn_beta     = (const float*)d_in[4];
  const float* wq = (const float*)d_in[5];
  const float* bq = (const float*)d_in[6];
  const float* wk = (const float*)d_in[7];
  const float* bk = (const float*)d_in[8];
  const float* wv = (const float*)d_in[9];
  const float* bv = (const float*)d_in[10];
  const float* wo = (const float*)d_in[11];
  const float* bo = (const float*)d_in[12];

  const size_t plane = (size_t)BATCH * CDIM * NPTS;  // 1 Mi elements
  const size_t wsz   = (size_t)CDIM * CDIM;          // 256 Ki elements

  char* p = (char*)d_ws;
  _Float16* hT   = (_Float16*)p;  p += plane * sizeof(_Float16);  // 2 MB
  _Float16* oT   = (_Float16*)p;  p += plane * sizeof(_Float16);  // 2 MB
  _Float16* wq16 = (_Float16*)p;  p += wsz * sizeof(_Float16);
  _Float16* wk16 = (_Float16*)p;  p += wsz * sizeof(_Float16);
  _Float16* wv16 = (_Float16*)p;  p += wsz * sizeof(_Float16);
  _Float16* wo16 = (_Float16*)p;  p += wsz * sizeof(_Float16);
  float* q = (float*)p;  p += plane * sizeof(float);              // 4 MB
  float* k = (float*)p;  p += plane * sizeof(float);
  float* v = (float*)p;  p += plane * sizeof(float);

  // 1) GroupNorm -> hT (f16, [b][n][c])
  dim3 gnGrid(NGROUPS, BATCH);
  gn_kernel<<<gnGrid, 256, 0, stream>>>(x, gn_gamma, gn_beta, hT);

  // 2) weights -> f16 (one-shot)
  int wn = (int)wsz, wblocks = (wn + 255) / 256;
  cvt_kernel<<<wblocks, 256, 0, stream>>>(wq, wq16, wn);
  cvt_kernel<<<wblocks, 256, 0, stream>>>(wk, wk16, wn);
  cvt_kernel<<<wblocks, 256, 0, stream>>>(wv, wv16, wn);
  cvt_kernel<<<wblocks, 256, 0, stream>>>(wo, wo16, wn);

  // 3) q/k/v projections (TDM-staged WMMA GEMMs)
  dim3 gGrid(CDIM / TM, NPTS / TN, BATCH);  // (4, 8, 2)
  gemm_kernel<<<gGrid, 256, 0, stream>>>(wq16, bq, hT, nullptr, q, CDIM, NPTS, CDIM);
  gemm_kernel<<<gGrid, 256, 0, stream>>>(wk16, bk, hT, nullptr, k, CDIM, NPTS, CDIM);
  gemm_kernel<<<gGrid, 256, 0, stream>>>(wv16, bv, hT, nullptr, v, CDIM, NPTS, CDIM);

  // 4) gathered attention -> oT (f16, permuted channel layout, transposed)
  attn_kernel<<<(BATCH * NHEADS * NPTS) / 256, 256, 0, stream>>>(
      q, k, v, att_idx, valid_mask, oT);

  // 5) output projection + residual (TDM-staged WMMA GEMM)
  gemm_kernel<<<gGrid, 256, 0, stream>>>(wo16, bo, oT, x, (float*)d_out, CDIM,
                                         NPTS, CDIM);
}
